// PseudoTokenGridEncoder_13417477833600
// MI455X (gfx1250) — matrix operand order repaired and checked by reading, not compiled, not run on hardware
//
#include <hip/hip_runtime.h>
#include <hip/hip_bf16.h>
#include <math.h>

// ---------------------------------------------------------------------------
// PseudoTokenGridEncoder, MI455X (gfx1250, wave32, WMMA).
//
// Bandwidth-bound (~220MB traffic after K/V fusion -> ~10us floor @23.3TB/s):
//  * No padded [B*S, P, E] buffer (reference pads with zeros; we use CSR-ish
//    per-bucket token lists, attention is permutation invariant per bucket).
//  * E x E projections are bf16 WMMA GEMMs, f32 accumulate
//    (v_wmma_f32_16x16x32_bf16). W staged in LDS pre-swizzled into per-lane
//    fragment order: one 32B vector load (2x ds_load_b128) per B fragment,
//    double-buffered so the next fragment load overlaps the current WMMA.
//  * K and V projections are FUSED (Wk+Wv both in LDS): activation rows are
//    read from HBM once instead of twice (saves ~84MB of traffic).
//  * Per-bucket attention: one wave per bucket, streaming online softmax
//    over bucket tokens + always-valid on-grid/fake token.
// ---------------------------------------------------------------------------

#define EDIM 128      // embedding dim (reference: E = 128)
#define MAXP 64       // per-bucket token cap (Poisson(4) tail << 64)

typedef __attribute__((ext_vector_type(16))) __bf16 bf16x16;
typedef __attribute__((ext_vector_type(8)))  float  floatx8;

// ---------------------------------------------------------------------------
__global__ void zero_i32_kernel(int* p, int n) {
    int i = blockIdx.x * blockDim.x + threadIdx.x;
    if (i < n) p[i] = 0;
}

// ---------------------------------------------------------------------------
// Nearest-grid-cell bucketize + per-bucket token lists.
// xc_on is a monotone meshgrid -> min/max are its first/last points.
__global__ void bucketize_kernel(const float* __restrict__ xc_off,
                                 const float* __restrict__ xc_on,
                                 int B, int U, int S, int G,
                                 int* __restrict__ counts,
                                 int* __restrict__ bucket_tok) {
    long i = (long)blockIdx.x * blockDim.x + threadIdx.x;   // over B*U
    if (i >= (long)B * U) return;
    int b = (int)(i / U);

    float x0 = xc_off[i * 2 + 0];
    float x1 = xc_off[i * 2 + 1];
    float xmin0 = xc_on[0], xmin1 = xc_on[1];
    float xmax0 = xc_on[(size_t)(S - 1) * 2 + 0];
    float xmax1 = xc_on[(size_t)(S - 1) * 2 + 1];
    float sp0 = (xmax0 - xmin0) / (float)(G - 1);
    float sp1 = (xmax1 - xmin1) / (float)(G - 1);

    float n0 = floorf((x0 - xmin0 + 0.5f * sp0) / sp0);
    float n1 = floorf((x1 - xmin1 + 0.5f * sp1) / sp1);
    n0 = fminf(fmaxf(n0, 0.0f), (float)(G - 1));
    n1 = fminf(fmaxf(n1, 0.0f), (float)(G - 1));

    int idx = (int)n0 * G + (int)n1;
    int seg = b * S + idx;
    int slot = atomicAdd(&counts[seg], 1);
    if (slot < MAXP) bucket_tok[(size_t)seg * MAXP + slot] = (int)i;
}

// ---------------------------------------------------------------------------
// Shared helpers for the WMMA GEMMs.
//
// LDS fragment-order swizzle (this code's contiguous k order, 16-bit B 32x16: lanes 0..15 carry
// K 0..15 of a K-block, lanes 16..31 carry K 16..31; column = nt*16+(lane&15)):
//   sW[(((kb*8 + nt)*32 + lane)*16 + e]  ==  bf16(W[k][col]),
//   k = kb*32 + (lane<16 ? 0 : 16) + e,  col = nt*16 + (lane&15)
// Each lane's fragment is one contiguous, 32B-aligned LDS vector.

static __device__ __forceinline__ void stage_weight_swizzled(
        __bf16* sW, const float* __restrict__ W, int tid) {
    for (int i = tid; i < EDIM * EDIM; i += 256) {
        const int k    = i >> 7;       // W row (K)
        const int col  = i & 127;      // W col (N)
        const int kb   = k >> 5;
        const int kr   = k & 31;
        const int nt   = col >> 4;
        const int lane = ((kr & 16) ? 16 : 0) + (col & 15);
        const int e    = kr & 15;
        sW[((((kb << 3) + nt) << 5) + lane) * 16 + e] = (__bf16)W[i];
    }
}

static __device__ __forceinline__ bf16x16 load_bfrag(
        const __bf16* sW, int kb, int nt, int lane) {
    return *(const bf16x16*)(sW + ((((kb << 3) + nt) << 5) + lane) * 16);
}

// A fragment (matches B swizzle): lane<16 -> K{kb*32+0..15},
// lane>=16 -> K{kb*32+16..31}: one contiguous 16-float run per lane.
static __device__ __forceinline__ bf16x16 load_afrag(
        const float* __restrict__ arow, int kb, int lane) {
    const int kbase = kb * 32 + ((lane < 16) ? 0 : 16);
    const float4 f0 = *(const float4*)(arow + kbase);
    const float4 f1 = *(const float4*)(arow + kbase + 4);
    const float4 f2 = *(const float4*)(arow + kbase + 8);
    const float4 f3 = *(const float4*)(arow + kbase + 12);
    bf16x16 a;
    a[0]  = (__bf16)f0.x; a[1]  = (__bf16)f0.y; a[2]  = (__bf16)f0.z; a[3]  = (__bf16)f0.w;
    a[4]  = (__bf16)f1.x; a[5]  = (__bf16)f1.y; a[6]  = (__bf16)f1.z; a[7]  = (__bf16)f1.w;
    a[8]  = (__bf16)f2.x; a[9]  = (__bf16)f2.y; a[10] = (__bf16)f2.z; a[11] = (__bf16)f2.w;
    a[12] = (__bf16)f3.x; a[13] = (__bf16)f3.y; a[14] = (__bf16)f3.z; a[15] = (__bf16)f3.w;
    return a;
}

// ---------------------------------------------------------------------------
// Single-weight GEMM: Out[M,128] = A[M,128] @ W[128,128].
// 256 threads = 8 waves; wave owns rows [m0, m0+16) and a 16x128 stripe.
// B fragments double-buffered so each ds load overlaps the previous WMMA.
// If (fake && *ignore_flag): every row of A is replaced by fake[0..127].
__global__ void gemm_bf16_wmma_kernel(const float* __restrict__ A, int M,
                                      const float* __restrict__ W,
                                      float* __restrict__ Out,
                                      const int* __restrict__ ignore_flag,
                                      const float* __restrict__ fake) {
    __shared__ __bf16 sW[EDIM * EDIM];   // 32 KB
    const int tid = threadIdx.x;
    stage_weight_swizzled(sW, W, tid);
    __syncthreads();

    const int wave = tid >> 5;
    const int lane = tid & 31;
    const int m0 = (blockIdx.x * 8 + wave) * 16;
    if (m0 >= M) return;   // wave-uniform: EXEC stays all-ones for WMMA

    const bool useFake = (fake != nullptr) && (ignore_flag != nullptr) &&
                         (*ignore_flag != 0);
    const int row = m0 + (lane & 15);
    const float* arow = useFake ? fake : (A + (size_t)row * EDIM);

    bf16x16 afrag[4];
#pragma unroll
    for (int kb = 0; kb < 4; ++kb) afrag[kb] = load_afrag(arow, kb, lane);

    floatx8 acc[8];
#pragma unroll
    for (int nt = 0; nt < 8; ++nt) acc[nt] = (floatx8){0,0,0,0,0,0,0,0};

    bf16x16 bcur = load_bfrag(sW, 0, 0, lane);
#pragma unroll
    for (int kb = 0; kb < 4; ++kb) {
#pragma unroll
        for (int nt = 0; nt < 8; ++nt) {
            const int nidx = kb * 8 + nt + 1;            // prefetch next frag
            bf16x16 bnext = bcur;
            if (nidx < 32) bnext = load_bfrag(sW, nidx >> 3, nidx & 7, lane);
            acc[nt] = __builtin_amdgcn_wmma_f32_16x16x32_bf16(
                false, afrag[kb], false, bcur, (short)0, acc[nt],
                false, false);
            bcur = bnext;
        }
    }

    // D layout: VGPR v, lane -> row m0 + v + (lane<16?0:8), col nt*16+(lane&15)
    const int rbase = m0 + ((lane < 16) ? 0 : 8);
#pragma unroll
    for (int nt = 0; nt < 8; ++nt) {
        const int col = nt * 16 + (lane & 15);
#pragma unroll
        for (int v = 0; v < 8; ++v)
            Out[(size_t)(rbase + v) * EDIM + col] = acc[nt][v];
    }
}

// ---------------------------------------------------------------------------
// Fused dual-weight GEMM: OutK = A @ Wk, OutV = A @ Wv.
// Activation rows are fetched from HBM ONCE for both projections.
__global__ void gemm_kv_bf16_wmma_kernel(const float* __restrict__ A, int M,
                                         const float* __restrict__ Wk,
                                         const float* __restrict__ Wv,
                                         float* __restrict__ OutK,
                                         float* __restrict__ OutV,
                                         const int* __restrict__ ignore_flag,
                                         const float* __restrict__ fake) {
    __shared__ __bf16 sWk[EDIM * EDIM];  // 32 KB
    __shared__ __bf16 sWv[EDIM * EDIM];  // 32 KB
    const int tid = threadIdx.x;
    stage_weight_swizzled(sWk, Wk, tid);
    stage_weight_swizzled(sWv, Wv, tid);
    __syncthreads();

    const int wave = tid >> 5;
    const int lane = tid & 31;
    const int m0 = (blockIdx.x * 8 + wave) * 16;
    if (m0 >= M) return;

    const bool useFake = (fake != nullptr) && (ignore_flag != nullptr) &&
                         (*ignore_flag != 0);
    const int row = m0 + (lane & 15);
    const float* arow = useFake ? fake : (A + (size_t)row * EDIM);

    bf16x16 afrag[4];
#pragma unroll
    for (int kb = 0; kb < 4; ++kb) afrag[kb] = load_afrag(arow, kb, lane);

    floatx8 acck[8], accv[8];
#pragma unroll
    for (int nt = 0; nt < 8; ++nt) {
        acck[nt] = (floatx8){0,0,0,0,0,0,0,0};
        accv[nt] = (floatx8){0,0,0,0,0,0,0,0};
    }

    bf16x16 bk = load_bfrag(sWk, 0, 0, lane);
    bf16x16 bv = load_bfrag(sWv, 0, 0, lane);
#pragma unroll
    for (int kb = 0; kb < 4; ++kb) {
#pragma unroll
        for (int nt = 0; nt < 8; ++nt) {
            const int nidx = kb * 8 + nt + 1;            // prefetch next pair
            bf16x16 bkn = bk, bvn = bv;
            if (nidx < 32) {
                bkn = load_bfrag(sWk, nidx >> 3, nidx & 7, lane);
                bvn = load_bfrag(sWv, nidx >> 3, nidx & 7, lane);
            }
            acck[nt] = __builtin_amdgcn_wmma_f32_16x16x32_bf16(
                false, afrag[kb], false, bk, (short)0, acck[nt], false, false);
            accv[nt] = __builtin_amdgcn_wmma_f32_16x16x32_bf16(
                false, afrag[kb], false, bv, (short)0, accv[nt], false, false);
            bk = bkn; bv = bvn;
        }
    }

    const int rbase = m0 + ((lane < 16) ? 0 : 8);
#pragma unroll
    for (int nt = 0; nt < 8; ++nt) {
        const int col = nt * 16 + (lane & 15);
#pragma unroll
        for (int v = 0; v < 8; ++v) {
            OutK[(size_t)(rbase + v) * EDIM + col] = acck[nt][v];
            OutV[(size_t)(rbase + v) * EDIM + col] = accv[nt][v];
        }
    }
}

// ---------------------------------------------------------------------------
// Streaming per-bucket attention. One wave per bucket n = b*S + s.
// Lane l owns output dims [4l, 4l+4); head h = l/4 spans lanes 4h..4h+3.
__global__ void bucket_attn_kernel(const float* __restrict__ Q,
                                   const float* __restrict__ Koff,
                                   const float* __restrict__ Voff,
                                   const float* __restrict__ Kon,
                                   const float* __restrict__ Von,
                                   const int*   __restrict__ counts,
                                   const int*   __restrict__ bucket_tok,
                                   float* __restrict__ attn_out,
                                   int NB, int S, float scale) {
    const int wave = threadIdx.x >> 5;
    const int lane = threadIdx.x & 31;
    const int n = blockIdx.x * (blockDim.x >> 5) + wave;
    if (n >= NB) return;
    const int s = n % S;

    float4 q = *(const float4*)(Q + (size_t)s * EDIM + lane * 4);
    q.x *= scale; q.y *= scale; q.z *= scale; q.w *= scale;

    float m = -3.0e38f, denom = 0.0f;
    float4 acc = {0.f, 0.f, 0.f, 0.f};

    int cnt = counts[n];
    if (cnt > MAXP) cnt = MAXP;

    for (int t = 0; t <= cnt; ++t) {            // t == cnt -> on-grid token
        const float* kp;
        const float* vp;
        if (t < cnt) {
            const int tok = bucket_tok[(size_t)n * MAXP + t];
            kp = Koff + (size_t)tok * EDIM;
            vp = Voff + (size_t)tok * EDIM;
        } else {
            kp = Kon + (size_t)n * EDIM;
            vp = Von + (size_t)n * EDIM;
        }
        const float4 kv = *(const float4*)(kp + lane * 4);
        float p = q.x * kv.x + q.y * kv.y + q.z * kv.z + q.w * kv.w;
        p += __shfl_xor(p, 1, 32);              // 4-lane head reduction
        p += __shfl_xor(p, 2, 32);

        const float nm   = fmaxf(m, p);
        const float corr = __expf(m - nm);      // first iter underflows to 0
        const float wgt  = __expf(p - nm);
        denom = denom * corr + wgt;

        const float4 vv = *(const float4*)(vp + lane * 4);
        acc.x = acc.x * corr + wgt * vv.x;
        acc.y = acc.y * corr + wgt * vv.y;
        acc.z = acc.z * corr + wgt * vv.z;
        acc.w = acc.w * corr + wgt * vv.w;
        m = nm;
    }

    const float inv = 1.0f / denom;             // denom > 0: on-grid token
    float4 o = {acc.x * inv, acc.y * inv, acc.z * inv, acc.w * inv};
    *(float4*)(attn_out + (size_t)n * EDIM + lane * 4) = o;
}

// ---------------------------------------------------------------------------
static inline size_t align256(size_t x) { return (x + 255) & ~(size_t)255; }

extern "C" void kernel_launch(void* const* d_in, const int* in_sizes, int n_in,
                              void* d_out, int out_size, void* d_ws, size_t ws_size,
                              hipStream_t stream) {
    // setup_inputs() order
    const float* xc_off   = (const float*)d_in[0];   // [B,U,2]
    const float* xc_on    = (const float*)d_in[1];   // [B,G,G,2]
    const float* zc_off   = (const float*)d_in[2];   // [B,U,E]
    const float* zc_on    = (const float*)d_in[3];   // [B,G,G,E]
    const float* latents  = (const float*)d_in[4];   // [S,E]
    const float* fake     = (const float*)d_in[5];   // [E]
    const float* Wq       = (const float*)d_in[6];
    const float* Wk       = (const float*)d_in[7];
    const float* Wv       = (const float*)d_in[8];
    const float* Wo       = (const float*)d_in[9];
    const int*   ignore   = (const int*)d_in[10];
    float*       out      = (float*)d_out;           // [B,G,G,E]

    const int E  = in_sizes[5];                      // 128
    const int S  = in_sizes[4] / E;                  // 4096
    const int B  = in_sizes[3] / (S * E);            // 8
    const int U  = in_sizes[2] / (B * E);            // 16384
    const int G  = (int)(sqrtf((float)S) + 0.5f);    // 64
    const int NB = B * S;                            // 32768
    const int H  = 8;                                // NUM_HEADS
    const float scale = 1.0f / sqrtf((float)(E / H));

    // workspace carve-up
    char* w = (char*)d_ws;
    size_t off = 0;
    int*   counts     = (int*)  (w + off); off = align256(off + (size_t)NB * 4);
    int*   bucket_tok = (int*)  (w + off); off = align256(off + (size_t)NB * MAXP * 4);
    float* Koff       = (float*)(w + off); off = align256(off + (size_t)B * U * E * 4);
    float* Voff       = (float*)(w + off); off = align256(off + (size_t)B * U * E * 4);
    float* Kon        = (float*)(w + off); off = align256(off + (size_t)NB * E * 4);
    float* Von        = (float*)(w + off); off = align256(off + (size_t)NB * E * 4);
    float* Qbuf       = (float*)(w + off); off = align256(off + (size_t)S * E * 4);
    float* attn       = (float*)(w + off); off = align256(off + (size_t)NB * E * 4);
    (void)ws_size; (void)n_in; (void)out_size;

    // 1) reset bucket counts
    zero_i32_kernel<<<(NB + 255) / 256, 256, 0, stream>>>(counts, NB);

    // 2) bucketize off-grid tokens
    {
        long total = (long)B * U;
        bucketize_kernel<<<(int)((total + 255) / 256), 256, 0, stream>>>(
            xc_off, xc_on, B, U, S, G, counts, bucket_tok);
    }

    // 3) projections (bf16 WMMA GEMMs); K+V fused -> activations read once
    gemm_bf16_wmma_kernel<<<(S + 127) / 128, 256, 0, stream>>>(
        latents, S, Wq, Qbuf, nullptr, nullptr);
    gemm_kv_bf16_wmma_kernel<<<(B * U + 127) / 128, 256, 0, stream>>>(
        zc_off, B * U, Wk, Wv, Koff, Voff, nullptr, nullptr);
    gemm_kv_bf16_wmma_kernel<<<(NB + 127) / 128, 256, 0, stream>>>(
        zc_on, NB, Wk, Wv, Kon, Von, ignore, fake);

    // 4) streaming bucket attention (8 waves/block -> 8 buckets/block)
    bucket_attn_kernel<<<(NB + 7) / 8, 256, 0, stream>>>(
        Qbuf, Koff, Voff, Kon, Von, counts, bucket_tok, attn, NB, S, scale);

    // 5) output projection
    gemm_bf16_wmma_kernel<<<(NB + 127) / 128, 256, 0, stream>>>(
        attn, NB, Wo, out, nullptr, nullptr);
}
